// SimpleAttention_18124761989121
// MI455X (gfx1250) — matrix operand order repaired and checked by reading, not compiled, not run on hardware
//
#include <hip/hip_runtime.h>
#include <hip/hip_bf16.h>

typedef __bf16 bf16;
typedef __attribute__((ext_vector_type(16))) __bf16 bf16x16;
typedef __attribute__((ext_vector_type(4)))  __bf16 bf16x4;
typedef __attribute__((ext_vector_type(8)))  float   f32x8;
typedef __attribute__((ext_vector_type(4)))  uint32_t u32x4;
typedef __attribute__((ext_vector_type(8)))  uint32_t u32x8;

#define BB   4
#define TT   2048
#define CC   1024
#define T3   3072
#define NH   16
#define HDIM 64
#define BT   8192   // B*T

// LDS byte offset of a __shared__ pointer (flat shared addr low 32 bits).
__device__ __forceinline__ uint32_t lds_off(const void* p) {
  return (uint32_t)(uintptr_t)p;
}
// Async global->LDS 16B copy (tracked by ASYNCcnt).
#define ASYNC_B128(ldsaddr, gaddr)                                             \
  asm volatile("global_load_async_to_lds_b128 %0, %1, off"                     \
               :: "v"(ldsaddr), "v"(gaddr) : "memory")
#define WAIT_ASYNC0()  asm volatile("s_wait_asynccnt 0x0" ::: "memory")
#define WAIT_DS0()     asm volatile("s_wait_dscnt 0x0" ::: "memory")

// ---- DPP16 butterfly reductions over 16-lane halves (no LDS) ----
// Masks {1,2,7,15} generate the 16-lane group: quad_perm[1,0,3,2],
// quad_perm[2,3,0,1], row_half_mirror, row_mirror.
template <int CTRL>
__device__ __forceinline__ float dpp_mov_f32(float x) {
  return __int_as_float(
      __builtin_amdgcn_mov_dpp(__float_as_int(x), CTRL, 0xf, 0xf, true));
}
__device__ __forceinline__ float red16_max(float x) {
  x = fmaxf(x, dpp_mov_f32<0xB1>(x));    // quad_perm [1,0,3,2]
  x = fmaxf(x, dpp_mov_f32<0x4E>(x));    // quad_perm [2,3,0,1]
  x = fmaxf(x, dpp_mov_f32<0x141>(x));   // row_half_mirror
  x = fmaxf(x, dpp_mov_f32<0x140>(x));   // row_mirror
  return x;
}
__device__ __forceinline__ float red16_sum(float x) {
  x += dpp_mov_f32<0xB1>(x);
  x += dpp_mov_f32<0x4E>(x);
  x += dpp_mov_f32<0x141>(x);
  x += dpp_mov_f32<0x140>(x);
  return x;
}

// ---- Tensor Data Mover: 2D tile load, D# per CDNA5 ISA ch.8 ----
// Loads tile_d1 rows of tile_d0 contiguous bf16 elements (row stride
// stride0 elements) into LDS at lds_byte, walking X then Y (row-major LDS).
__device__ __forceinline__ void tdm_load_2d(uint32_t lds_byte, const bf16* gp,
                                            uint32_t tile_d0, uint32_t tile_d1,
                                            uint64_t stride0) {
  uint64_t ga = (uint64_t)(uintptr_t)gp;
  u32x4 g0;
  g0[0] = 1u;                                   // count=1, user descriptor
  g0[1] = lds_byte;                             // lds_addr
  g0[2] = (uint32_t)ga;                         // global_addr[31:0]
  g0[3] = (uint32_t)((ga >> 32) & 0x01FFFFFFu)  // global_addr[56:32]
        | (2u << 30);                           // type = 2 ("image")
  u32x8 g1;
  g1[0] = 1u << 16;                             // data_size = 1 -> 2 bytes
  g1[1] = (tile_d0 & 0xFFFFu) << 16;            // tensor_dim0[15:0] (== tile)
  g1[2] = (tile_d0 >> 16) | ((tile_d1 & 0xFFFFu) << 16);  // dim0 hi | dim1 lo
  g1[3] = (tile_d1 >> 16) | (tile_d0 << 16);    // dim1 hi | tile_dim0
  g1[4] = tile_d1 & 0xFFFFu;                    // tile_dim1 (tile_dim2 = 0)
  g1[5] = (uint32_t)stride0;                    // tensor_dim0_stride[31:0]
  g1[6] = (uint32_t)(stride0 >> 32) & 0xFFFFu;  // stride0[47:32] (dim1_stride=0)
  g1[7] = 0u;
  asm volatile("tensor_load_to_lds %0, %1, null, null"
               :: "s"(g0), "s"(g1) : "memory");
}
#define WAIT_TENSOR0() __builtin_amdgcn_s_wait_tensorcnt(0)

// ---------------- f32 -> bf16 convert (x only) ----------------
__global__ void cvt_f32_bf16(const float* __restrict__ src, bf16* __restrict__ dst, int n4) {
  int i = blockIdx.x * blockDim.x + threadIdx.x;
  if (i >= n4) return;
  const float4 f = ((const float4*)src)[i];
  bf16x4 o;
  o[0] = (bf16)f.x; o[1] = (bf16)f.y; o[2] = (bf16)f.z; o[3] = (bf16)f.w;
  ((bf16x4*)dst)[i] = o;
}

// ------------- f32 [K,N] -> bf16 [N,K] tile transpose -------------
__global__ __launch_bounds__(256) void transpose_cvt(
    const float* __restrict__ W, bf16* __restrict__ Wt, int K, int N) {
  __shared__ float tile[32][33];
  const int k0 = blockIdx.x * 32, n0 = blockIdx.y * 32;
  const int tx = threadIdx.x, ty = threadIdx.y;  // 32 x 8
#pragma unroll
  for (int i = 0; i < 4; ++i)
    tile[ty + 8 * i][tx] = W[(size_t)(k0 + ty + 8 * i) * N + n0 + tx];
  __syncthreads();
#pragma unroll
  for (int i = 0; i < 4; ++i)
    Wt[(size_t)(n0 + ty + 8 * i) * K + k0 + tx] = (bf16)tile[tx][ty + 8 * i];
}

// ---------------- tiled bf16 GEMM: C = A[MxK] * Wt[NxK]^T + bias ------
// 256 threads = 8 waves (4M x 2N). Block tile 128x128, K-step 32.
// Async double-buffered LDS staging; all fragment reads contiguous.
// If vt != nullptr, columns [2C,3C) are written transposed to vt[b,h,d,t].
template <int OUT_BF16>
__global__ __launch_bounds__(256) void gemm_bf16(
    const bf16* __restrict__ A, const bf16* __restrict__ Wt,
    const float* __restrict__ bias,
    bf16* __restrict__ Cb, float* __restrict__ Cf, bf16* __restrict__ vt,
    int M, int N, int K)
{
  __shared__ __align__(16) bf16 As[2][128 * 32];  // [m][k]
  __shared__ __align__(16) bf16 Bs[2][128 * 32];  // [n][k]

  const int tid  = threadIdx.x;
  const int lane = tid & 31;
  const int wave = tid >> 5;
  const int lm   = lane & 15;
  const int lg   = lane >> 4;
  const int wm   = wave & 3;
  const int wn   = wave >> 2;
  const int mBase = blockIdx.x * 128;
  const int nBase = blockIdx.y * 128;

  auto issue = [&](int buf, int k0) {
#pragma unroll
    for (int it = 0; it < 2; ++it) {
      int c  = tid + it * 256;          // 512 chunks of 8 bf16
      int r  = c >> 2, c8 = (c & 3) * 8;
      uint64_t ga = (uint64_t)(uintptr_t)(A  + (size_t)(mBase + r) * K + k0 + c8);
      uint64_t gb = (uint64_t)(uintptr_t)(Wt + (size_t)(nBase + r) * K + k0 + c8);
      uint32_t la = lds_off(&As[buf][0]) + c * 16;
      uint32_t lb = lds_off(&Bs[buf][0]) + c * 16;
      ASYNC_B128(la, ga);
      ASYNC_B128(lb, gb);
    }
  };

  f32x8 acc[2][4];
#pragma unroll
  for (int i = 0; i < 2; ++i)
#pragma unroll
    for (int j = 0; j < 4; ++j)
#pragma unroll
      for (int v = 0; v < 8; ++v) acc[i][j][v] = 0.0f;

  const int nk = K / 32;
  issue(0, 0);
  for (int kt = 0; kt < nk; ++kt) {
    const int buf = kt & 1;
    WAIT_ASYNC0();
    __syncthreads();                    // tile kt resident for all waves

    bf16x16 af[2];
#pragma unroll
    for (int i = 0; i < 2; ++i) {
      const bf16* p = &As[buf][(wm * 32 + i * 16 + lm) * 32];
#pragma unroll
      for (int e = 0; e < 8; ++e) { af[i][e] = p[lg * 16 + e]; af[i][8 + e] = p[lg * 16 + 8 + e]; }
    }
    bf16x16 bfm[4];
#pragma unroll
    for (int j = 0; j < 4; ++j) {
      const bf16* p = &Bs[buf][(wn * 64 + j * 16 + lm) * 32 + lg * 16];
#pragma unroll
      for (int e = 0; e < 16; ++e) bfm[j][e] = p[e];
    }
    WAIT_DS0();
    __syncthreads();                    // all waves done reading this buffer pair
    if (kt + 1 < nk) issue(buf ^ 1, (kt + 1) * 32);  // DMA overlaps WMMA below

#pragma unroll
    for (int i = 0; i < 2; ++i)
#pragma unroll
      for (int j = 0; j < 4; ++j)
        acc[i][j] = __builtin_amdgcn_wmma_f32_16x16x32_bf16(
            false, af[i], false, bfm[j], (short)0, acc[i][j], false, false);
  }

  // epilogue: C layout rows = v + 8*lg at fixed col = lm
#pragma unroll
  for (int i = 0; i < 2; ++i) {
    int rb = mBase + wm * 32 + i * 16 + 8 * lg;
#pragma unroll
    for (int j = 0; j < 4; ++j) {
      int col = nBase + wn * 64 + j * 16 + lm;
      float bia = bias[col];
      if (OUT_BF16) {
        if (vt != nullptr && col >= 2 * CC) {
          // V columns -> transposed vt[(b*H+h)*HD+d][t]; rows are consecutive t
          int hd = col - 2 * CC;
          int bidx = rb / TT, tb = rb % TT;
          bf16* vp = vt + ((size_t)bidx * CC + hd) * TT + tb;
#pragma unroll
          for (int v = 0; v < 8; ++v) vp[v] = (bf16)(acc[i][j][v] + bia);
        } else {
#pragma unroll
          for (int v = 0; v < 8; ++v)
            Cb[(size_t)(rb + v) * N + col] = (bf16)(acc[i][j][v] + bia);
        }
      } else {
#pragma unroll
        for (int v = 0; v < 8; ++v)
          Cf[(size_t)(rb + v) * N + col] = acc[i][j][v] + bia;
      }
    }
  }
}

// ---------------- flash attention (causal), bf16 WMMA + TDM ----------------
// grid: (T/128, B*H). 8 waves; each wave owns 16 q rows. KV tiles of 32.
// K/V tiles staged by the Tensor Data Mover (wave 0 issues, TENSORcnt sync).
// K staged [kv][d], V staged (pre-transposed) [d][kv]: all frag reads contiguous.
__global__ __launch_bounds__(256) void attn_kernel(
    const bf16* __restrict__ qkv, const bf16* __restrict__ vt,
    bf16* __restrict__ y)
{
  __shared__ __align__(16) bf16 Ks[2][32 * 64];
  __shared__ __align__(16) bf16 Vs[2][64 * 32];
  __shared__ __align__(16) bf16 Ps[8 * 16 * 32];

  const int tid  = threadIdx.x;
  const int lane = tid & 31;
  const int wave = tid >> 5;
  const int lm   = lane & 15;
  const int lg   = lane >> 4;
  const int qbase = blockIdx.x * 128;
  const int bh = blockIdx.y;
  const int b = bh / NH, h = bh % NH;

  const bf16* qp = qkv + (size_t)(b * TT + qbase + wave * 16 + lm) * T3 + h * HDIM;
  bf16x16 a0, a1;
#pragma unroll
  for (int e = 0; e < 8; ++e) {
    a0[e] = qp[lg * 16 + e];       a0[8 + e] = qp[lg * 16 + 8 + e];
    a1[e] = qp[32 + lg * 16 + e];  a1[8 + e] = qp[40 + lg * 16 + e];
  }

  const bf16* kbase = qkv + (size_t)(b * TT) * T3 + CC + h * HDIM;
  const bf16* vbase = vt + (size_t)bh * HDIM * TT;  // [d][t]

  // TDM staging: K tile = 32 rows x 64 elems (stride T3), V tile = 64 x 32 (stride TT)
  auto issue = [&](int buf, int kv0) {
    tdm_load_2d(lds_off(&Ks[buf][0]), kbase + (size_t)kv0 * T3, HDIM, 32, T3);
    tdm_load_2d(lds_off(&Vs[buf][0]), vbase + kv0,              32, HDIM, TT);
  };

  f32x8 o[4];
  float Mrow[8], Lrow[8];
#pragma unroll
  for (int td = 0; td < 4; ++td)
#pragma unroll
    for (int v = 0; v < 8; ++v) o[td][v] = 0.0f;
#pragma unroll
  for (int v = 0; v < 8; ++v) { Mrow[v] = -1e30f; Lrow[v] = 0.0f; }

  const int nkv = (qbase + 128) / 32;
  if (wave == 0) issue(0, 0);

  for (int kt = 0; kt < nkv; ++kt) {
    const int buf = kt & 1;
    const int kv0 = kt * 32;
    WAIT_TENSOR0();                     // only wave 0 has TENSORcnt != 0
    __syncthreads();

    // wave-uniform causal classification of this tile for this wave's rows
    const int rmin = qbase + wave * 16;
    const bool active    = kv0 <= rmin + 15;       // not fully above diagonal
    const bool need_mask = active && (kv0 + 31 > rmin);

    bf16x16 ap, bv[4];
    if (active) {
      // S = Q @ K^T  (B cols = K rows -> contiguous)
      f32x8 s[2];
#pragma unroll
      for (int t = 0; t < 2; ++t) {
#pragma unroll
        for (int v = 0; v < 8; ++v) s[t][v] = 0.0f;
        bf16x16 b0, b1;
        const bf16* p = &Ks[buf][(t * 16 + lm) * 64 + lg * 16];
#pragma unroll
        for (int e = 0; e < 16; ++e) { b0[e] = p[e]; b1[e] = p[32 + e]; }
        s[t] = __builtin_amdgcn_wmma_f32_16x16x32_bf16(false, a0, false, b0, (short)0, s[t], false, false);
        s[t] = __builtin_amdgcn_wmma_f32_16x16x32_bf16(false, a1, false, b1, (short)0, s[t], false, false);
      }

      // online softmax per row (rows = v + 8*lg; DPP16 butterfly reductions)
      float p0[8], p1[8];
#pragma unroll
      for (int v = 0; v < 8; ++v) {
        float s0 = s[0][v] * 0.125f;
        float s1 = s[1][v] * 0.125f;
        if (need_mask) {
          int qrow = rmin + v + 8 * lg;
          if (kv0 + lm > qrow)      s0 = -1e30f;
          if (kv0 + 16 + lm > qrow) s1 = -1e30f;
        }
        float mx = red16_max(fmaxf(s0, s1));
        float mnew = fmaxf(Mrow[v], mx);
        float alpha = __expf(Mrow[v] - mnew);
        Mrow[v] = mnew;
        float e0 = __expf(s0 - mnew), e1 = __expf(s1 - mnew);
        float sum = red16_sum(e0 + e1);
        Lrow[v] = Lrow[v] * alpha + sum;
#pragma unroll
        for (int td = 0; td < 4; ++td) o[td][v] *= alpha;
        p0[v] = e0; p1[v] = e1;
      }

      // C-layout -> A-layout via wave-private LDS (same-wave DS in-order)
      bf16* Pw = Ps + wave * (16 * 32);
#pragma unroll
      for (int v = 0; v < 8; ++v) {
        int rl = v + 8 * lg;
        Pw[rl * 32 + lm]      = (bf16)p0[v];
        Pw[rl * 32 + 16 + lm] = (bf16)p1[v];
      }
      {
        const bf16* p = Pw + lm * 32;
#pragma unroll
        for (int e = 0; e < 8; ++e) { ap[e] = p[lg * 16 + e]; ap[8 + e] = p[lg * 16 + 8 + e]; }
      }
      // preload V fragments (Vs is [d][kv] -> contiguous)
#pragma unroll
      for (int td = 0; td < 4; ++td) {
        const bf16* p = &Vs[buf][(td * 16 + lm) * 32 + lg * 16];
#pragma unroll
        for (int e = 0; e < 16; ++e) bv[td][e] = p[e];
      }
    }

    WAIT_DS0();
    __syncthreads();                     // everyone done reading this buffer
    if (wave == 0 && kt + 1 < nkv) issue(buf ^ 1, kv0 + 32);  // DMA overlaps WMMAs

    if (active) {
#pragma unroll
      for (int td = 0; td < 4; ++td)
        o[td] = __builtin_amdgcn_wmma_f32_16x16x32_bf16(
            false, ap, false, bv[td], (short)0, o[td], false, false);
    }
  }

  // normalize + store y in [B*T, C]
#pragma unroll
  for (int v = 0; v < 8; ++v) {
    float inv = 1.0f / Lrow[v];
    int qrow = qbase + wave * 16 + v + 8 * lg;
    bf16* yp = y + (size_t)(b * TT + qrow) * CC + h * HDIM;
#pragma unroll
    for (int td = 0; td < 4; ++td)
      yp[td * 16 + lm] = (bf16)(o[td][v] * inv);
  }
}

// ---------------- host side ----------------
extern "C" void kernel_launch(void* const* d_in, const int* in_sizes, int n_in,
                              void* d_out, int out_size, void* d_ws, size_t ws_size,
                              hipStream_t stream) {
  const float* x  = (const float*)d_in[0];
  const float* Wa = (const float*)d_in[1];
  const float* ba = (const float*)d_in[2];
  const float* Wp = (const float*)d_in[3];
  const float* bp = (const float*)d_in[4];
  float* out = (float*)d_out;

  bf16* xb   = (bf16*)d_ws;                       // [8192][1024]
  bf16* wab  = xb   + (size_t)BT * CC;            // [3072][1024] (transposed)
  bf16* wpb  = wab  + (size_t)CC * T3;            // [1024][1024] (transposed)
  bf16* qkvb = wpb  + (size_t)CC * CC;            // [8192][3072] (q,k used)
  bf16* yb   = qkvb + (size_t)BT * T3;            // [8192][1024]
  bf16* vtb  = yb   + (size_t)BT * CC;            // [B*H*HD][T]

  {
    int n4 = (BT * CC) / 4;
    cvt_f32_bf16<<<(n4 + 255) / 256, 256, 0, stream>>>(x, xb, n4);
  }
  transpose_cvt<<<dim3(CC / 32, T3 / 32), dim3(32, 8), 0, stream>>>(Wa, wab, CC, T3);
  transpose_cvt<<<dim3(CC / 32, CC / 32), dim3(32, 8), 0, stream>>>(Wp, wpb, CC, CC);

  // qkv = x @ W_attn + b_attn  (bf16; V written transposed to vtb)
  gemm_bf16<1><<<dim3(BT / 128, T3 / 128), 256, 0, stream>>>(
      xb, wab, ba, qkvb, nullptr, vtb, BT, T3, CC);

  // causal flash attention (TDM-staged K/V)
  attn_kernel<<<dim3(TT / 128, BB * NH), 256, 0, stream>>>(qkvb, vtb, yb);

  // out = y @ W_proj + b_proj  (f32)
  gemm_bf16<0><<<dim3(BT / 128, CC / 128), 256, 0, stream>>>(
      yb, wpb, bp, nullptr, out, nullptr, BT, CC, CC);
}
